// DDDBlock_7799660609573
// MI455X (gfx1250) — matrix operand order repaired and checked by reading, not compiled, not run on hardware
//
#include <hip/hip_runtime.h>
#include <hip/hip_bf16.h>

// ---------------- problem constants ----------------
#define BN 2
#define CC 64
#define HH 192
#define WW 192
#define HWP (HH * WW)          // 36864
#define TPR (WW / 16)          // 12 tiles per row
#define NBLK (BN * HH * TPR)   // 4608 tile blocks

typedef __attribute__((ext_vector_type(8)))  float          v8f;
typedef __attribute__((ext_vector_type(16))) __bf16         v16bf;
typedef __attribute__((ext_vector_type(16))) unsigned short v16u16;
typedef __attribute__((ext_vector_type(8)))  unsigned short u16x8;
typedef __attribute__((ext_vector_type(4)))  int            v4i;

typedef __attribute__((address_space(1))) v4i* as1_v4i_ptr;
typedef __attribute__((address_space(3))) v4i* as3_v4i_ptr;

__device__ __forceinline__ unsigned short f2bf(float f) {
    unsigned u = __builtin_bit_cast(unsigned, f);
    u += 0x7FFFu + ((u >> 16) & 1u);      // round-to-nearest-even
    return (unsigned short)(u >> 16);
}
__device__ __forceinline__ float b2f(unsigned short u) {
    return __builtin_bit_cast(float, (unsigned)u << 16);
}

// ---- CDNA5 async global->LDS copy (16B per lane), ASYNCcnt tracked ----
__device__ __forceinline__ void async_ld16(const unsigned short* g, unsigned short* l) {
#if __has_builtin(__builtin_amdgcn_global_load_async_to_lds_b128)
    __builtin_amdgcn_global_load_async_to_lds_b128(
        (as1_v4i_ptr)(void*)g, (as3_v4i_ptr)(void*)l, 0, 0);
#else
    unsigned lo = (unsigned)(unsigned long)(__attribute__((address_space(3))) unsigned short*)l;
    asm volatile("global_load_async_to_lds_b128 %0, %1, off"
                 :: "v"(lo), "v"(g) : "memory");
#endif
}
__device__ __forceinline__ void wait_async0() {
#if __has_builtin(__builtin_amdgcn_s_wait_asynccnt)
    __builtin_amdgcn_s_wait_asynccnt(0);
#else
    asm volatile("s_wait_asynccnt 0x0" ::: "memory");
#endif
}

// ---------------- 1) channel LayerNorm: NCHW f32 -> NHWC bf16 ----------------
__global__ void ln_kernel(const float* __restrict__ x,
                          const float* __restrict__ lw,
                          const float* __restrict__ lb,
                          unsigned short* __restrict__ xn16) {
    long i = (long)blockIdx.x * blockDim.x + threadIdx.x;   // pixel over B*H*W
    if (i >= (long)BN * HWP) return;
    int b = (int)(i / HWP);
    int pix = (int)(i - (long)b * HWP);
    const float* px = x + (long)b * CC * HWP + pix;
    float s = 0.f, ss = 0.f;
    for (int c = 0; c < CC; ++c) { float v = px[(long)c * HWP]; s += v; ss += v * v; }
    float mu  = s * (1.f / CC);
    float var = ss * (1.f / CC) - mu * mu;      // biased variance
    float inv = rsqrtf(var + 1e-5f);
    unsigned short* po = xn16 + ((long)b * HWP + pix) * CC;
    for (int c = 0; c < CC; ++c)
        po[c] = f2bf((px[(long)c * HWP] - mu) * inv * lw[c] + lb[c]);
}

// ---------------- 2) weight packing into WMMA-B fragments ----------------
// dst layout: [s=0..17][n=0..Npad-1][kloc=0..31] bf16
// k-step s: tap t = s>>1, channels c = (s&1)*32 + kloc
__global__ void pack64_kernel(const float* __restrict__ w, unsigned short* __restrict__ dst) {
    int i = blockIdx.x * blockDim.x + threadIdx.x;
    if (i >= 18 * 64 * 32) return;
    int kloc = i & 31;
    int n    = (i >> 5) & 63;
    int s    = i >> 11;
    int t = s >> 1;
    int c = (s & 1) * 32 + kloc;
    dst[i] = f2bf(w[((long)n * CC + c) * 9 + t]);   // OIHW(3x3) flattened
}

__global__ void pack_om_kernel(const float* __restrict__ ow, const float* __restrict__ mw,
                               const float* __restrict__ ob, const float* __restrict__ mb,
                               unsigned short* __restrict__ dst, float* __restrict__ biasdst) {
    int i = blockIdx.x * blockDim.x + threadIdx.x;
    if (i >= 18 * 32 * 32) return;
    int kloc = i & 31;
    int n    = (i >> 5) & 31;
    int s    = i >> 10;
    int t = s >> 1;
    int c = (s & 1) * 32 + kloc;
    float v = 0.f;
    if (n < 18)      v = ow[((long)n * CC + c) * 9 + t];
    else if (n < 27) v = mw[((long)(n - 18) * CC + c) * 9 + t];
    dst[i] = f2bf(v);
    if (i < 32) {
        float bv = 0.f;
        if (i < 18)      bv = ob[i];
        else if (i < 27) bv = mb[i - 18];
        biasdst[i] = bv;
    }
}

// ---------------- 3) implicit-GEMM 3x3 conv via WMMA ----------------
// Input: NHWC bf16. One wave computes a 16-pixel x Npad tile. K = 576 in 18 steps
// (fully unrolled so the scheduler can hoist step s+1 loads over step s WMMAs).
template <int NT, bool BLEND>
__global__ __launch_bounds__(32) void conv3x3_wmma(
        const unsigned short* __restrict__ in16, // [B][H][W][64] bf16
        const unsigned short* __restrict__ Bp,   // packed weights
        const float* __restrict__ bias,          // [Npad]
        float* __restrict__ out,                 // [B][Npad][H][W] f32
        const float* __restrict__ pm,            // pre_mask (BLEND only)
        const float* __restrict__ resid) {       // residual x (BLEND only)
    constexpr int Npad = NT * 16;
    __shared__ unsigned short tile[54 * 72];
    int bid = blockIdx.x;
    int x0  = (bid % TPR) * 16;
    int y   = (bid / TPR) % HH;
    int b   = bid / (TPR * HH);
    int lane = threadIdx.x;

    // stage 3x18 halo x 64ch via async 16B chunks; zero-fill out-of-bounds
    const unsigned short* gb = in16 + (long)b * HWP * CC;
    for (int q = lane; q < 54 * 8; q += 32) {
        int p  = q >> 3;
        int ck = (q & 7) * 8;
        int r  = p / 18, xi = p - r * 18;
        int yy = y + r - 1, xx = x0 + xi - 1;
        if (yy >= 0 && yy < HH && xx >= 0 && xx < WW) {
            async_ld16(gb + ((long)yy * WW + xx) * CC + ck, &tile[p * 72 + ck]);
        } else {
            u16x8 z = {};
            *(u16x8*)&tile[p * 72 + ck] = z;
        }
    }
    wait_async0();
    __syncthreads();

    int m = lane & 15, h = lane >> 4;
    v8f acc[NT] = {};

    #pragma unroll
    for (int s = 0; s < 18; ++s) {
        int t  = s >> 1;
        int c0 = (s & 1) * 32;
        int ty = t / 3, tx = t - ty * 3;
        // A fragment: lane holds row M=m; element e -> K=c0+16h+e
        int base = (ty * 18 + m + tx) * 72 + c0 + 16 * h;
        u16x8 ch0 = *(const u16x8*)&tile[base];
        u16x8 ch1 = *(const u16x8*)&tile[base + 8];
        v16u16 au;
        #pragma unroll
        for (int j = 0; j < 8; ++j) { au[j] = ch0[j]; au[8 + j] = ch1[j]; }
        v16bf a = __builtin_bit_cast(v16bf, au);

        const unsigned short* bpS = Bp + ((long)s * Npad + m) * 32 + h * 16;
        v16u16 bu[NT];
        #pragma unroll
        for (int nt = 0; nt < NT; ++nt) bu[nt] = *(const v16u16*)(bpS + nt * 16 * 32);
        #pragma unroll
        for (int nt = 0; nt < NT; ++nt) {
            acc[nt] = __builtin_amdgcn_wmma_f32_16x16x32_bf16(
                false, a, false, __builtin_bit_cast(v16bf, bu[nt]), (short)0, acc[nt], false, false);
        }
    }

    // D layout: VGPR r -> pixel x0 + r + 8h, channel nt*16 + m
    #pragma unroll
    for (int nt = 0; nt < NT; ++nt) {
        int ch = nt * 16 + m;
        float bv = bias[ch];
        #pragma unroll
        for (int r = 0; r < 8; ++r) {
            int xp = x0 + r + 8 * h;
            long idx = ((long)(b * Npad + ch) * HH + y) * WW + xp;
            float v = acc[nt][r] + bv;
            if constexpr (BLEND) {
                float pmv = pm[((long)b * HH + y) * WW + xp];
                v = v * pmv + resid[idx] * (1.f - pmv);
            }
            out[idx] = v;
        }
    }
}

// ---------------- 4) modulated deformable conv via WMMA ----------------
// A = bilinear-gathered, mask-modulated xn (NHWC bf16): each corner's 8 channels
// is one contiguous 16B load. Corner validity handled branchless (clamped address
// + cndmask'd weight). Output x_def as NHWC bf16.
__global__ __launch_bounds__(32) void deform_wmma_kernel(
        const unsigned short* __restrict__ xn16, // [B][H][W][64] bf16
        const float* __restrict__ om,            // [B][32][H][W] f32: 0..17 off, 18..26 mask
        const unsigned short* __restrict__ Bp,   // packed def weights [18][64][32]
        const float* __restrict__ bias,          // def_b[64]
        unsigned short* __restrict__ out16) {    // x_def [B][H][W][64] bf16
    __shared__ float sw[144 * 4];
    __shared__ int   sij[144 * 2];
    int bid = blockIdx.x;
    int x0  = (bid % TPR) * 16;
    int y   = (bid / TPR) % HH;
    int b   = bid / (TPR * HH);
    int lane = threadIdx.x;

    for (int p = lane; p < 144; p += 32) {       // p = t*16 + m
        int m = p & 15, t = p >> 4;
        int ty = t / 3, tx = t - ty * 3;
        int xx = x0 + m;
        const float* omp = om + (long)b * 32 * HWP + (long)y * WW + xx;
        float dy = omp[(long)(2 * t) * HWP];
        float dx = omp[(long)(2 * t + 1) * HWP];
        float mk = omp[(long)(18 + t) * HWP];
        mk = 1.f / (1.f + expf(-mk));            // sigmoid
        float py = dy + (float)(y - 1 + ty);
        float px = dx + (float)(xx - 1 + tx);
        float fy = floorf(py), fx = floorf(px);
        float ly = py - fy, lx = px - fx;
        sij[2 * p]     = (int)fy;
        sij[2 * p + 1] = (int)fx;
        sw[4 * p]     = (1.f - ly) * (1.f - lx) * mk;
        sw[4 * p + 1] = (1.f - ly) * lx * mk;
        sw[4 * p + 2] = ly * (1.f - lx) * mk;
        sw[4 * p + 3] = ly * lx * mk;
    }
    __syncthreads();

    int m = lane & 15, h = lane >> 4;
    v8f acc[4] = {};
    const unsigned short* xb = xn16 + (long)b * HWP * CC;

    #pragma unroll 2
    for (int s = 0; s < 18; ++s) {
        int t  = s >> 1;
        int c0 = (s & 1) * 32;
        int p = t * 16 + m;
        int iy = sij[2 * p], ix = sij[2 * p + 1];
        float wgt[4] = { sw[4 * p], sw[4 * p + 1], sw[4 * p + 2], sw[4 * p + 3] };
        long off[4];
        #pragma unroll
        for (int k = 0; k < 4; ++k) {
            int yy = iy + (k >> 1);
            int xx = ix + (k & 1);
            bool ok = ((unsigned)yy < (unsigned)HH) & ((unsigned)xx < (unsigned)WW);
            int yc = yy < 0 ? 0 : (yy > HH - 1 ? HH - 1 : yy);
            int xc = xx < 0 ? 0 : (xx > WW - 1 ? WW - 1 : xx);
            off[k] = ((long)yc * WW + xc) * CC;       // always valid address
            wgt[k] = ok ? wgt[k] : 0.f;               // cndmask, no branch
        }

        int cA = c0 + 16 * h;        // channels of elements e=0..7
        int cB = c0 + 16 * h + 8;    // channels of elements e=8..15
        u16x8 a00 = *(const u16x8*)(xb + off[0] + cA), b00 = *(const u16x8*)(xb + off[0] + cB);
        u16x8 a01 = *(const u16x8*)(xb + off[1] + cA), b01 = *(const u16x8*)(xb + off[1] + cB);
        u16x8 a10 = *(const u16x8*)(xb + off[2] + cA), b10 = *(const u16x8*)(xb + off[2] + cB);
        u16x8 a11 = *(const u16x8*)(xb + off[3] + cA), b11 = *(const u16x8*)(xb + off[3] + cB);

        v16bf a;
        #pragma unroll
        for (int j = 0; j < 8; ++j) {
            float v0 = wgt[0] * b2f(a00[j]) + wgt[1] * b2f(a01[j]) + wgt[2] * b2f(a10[j]) + wgt[3] * b2f(a11[j]);
            float v1 = wgt[0] * b2f(b00[j]) + wgt[1] * b2f(b01[j]) + wgt[2] * b2f(b10[j]) + wgt[3] * b2f(b11[j]);
            a[j]     = (__bf16)v0;    // native v_cvt to bf16 (RNE)
            a[8 + j] = (__bf16)v1;
        }

        const unsigned short* bpS = Bp + ((long)s * 64 + m) * 32 + h * 16;
        v16u16 bu[4];
        #pragma unroll
        for (int nt = 0; nt < 4; ++nt) bu[nt] = *(const v16u16*)(bpS + nt * 16 * 32);
        #pragma unroll
        for (int nt = 0; nt < 4; ++nt) {
            acc[nt] = __builtin_amdgcn_wmma_f32_16x16x32_bf16(
                false, a, false, __builtin_bit_cast(v16bf, bu[nt]), (short)0, acc[nt], false, false);
        }
    }

    #pragma unroll
    for (int nt = 0; nt < 4; ++nt) {
        int ch = nt * 16 + m;
        float bv = bias[ch];
        #pragma unroll
        for (int r = 0; r < 8; ++r) {
            int xp = x0 + r + 8 * h;
            out16[((long)b * HWP + (long)y * WW + xp) * CC + ch] = f2bf(acc[nt][r] + bv);
        }
    }
}

// ---------------- launch ----------------
extern "C" void kernel_launch(void* const* d_in, const int* in_sizes, int n_in,
                              void* d_out, int out_size, void* d_ws, size_t ws_size,
                              hipStream_t stream) {
    const float* x        = (const float*)d_in[0];
    const float* pre_mask = (const float*)d_in[1];
    const float* ln_w     = (const float*)d_in[2];
    const float* ln_b     = (const float*)d_in[3];
    const float* off_w    = (const float*)d_in[4];
    const float* off_b    = (const float*)d_in[5];
    const float* mask_w   = (const float*)d_in[6];
    const float* mask_b   = (const float*)d_in[7];
    const float* def_w    = (const float*)d_in[8];
    const float* def_b    = (const float*)d_in[9];
    const float* fus_w    = (const float*)d_in[10];
    const float* fus_b    = (const float*)d_in[11];
    float* out = (float*)d_out;

    char* ws = (char*)d_ws;
    const size_t SZ_XN16   = (size_t)BN * HWP * CC * sizeof(unsigned short); // 9.4 MB
    const size_t SZ_OM     = (size_t)BN * 32 * HWP * sizeof(float);          // 9.4 MB
    const size_t SZ_XDEF16 = SZ_XN16;                                        // 9.4 MB
    unsigned short* xn16   = (unsigned short*)(ws);
    float*          om     = (float*)(ws + SZ_XN16);
    unsigned short* xdef16 = (unsigned short*)(ws + SZ_XN16 + SZ_OM);
    unsigned short* bp_om  = (unsigned short*)(ws + SZ_XN16 + SZ_OM + SZ_XDEF16);
    unsigned short* bp_def = (unsigned short*)((char*)bp_om  + 18 * 32 * 32 * sizeof(unsigned short));
    unsigned short* bp_fus = (unsigned short*)((char*)bp_def + 18 * 64 * 32 * sizeof(unsigned short));
    float* bias_om = (float*)((char*)bp_fus + 18 * 64 * 32 * sizeof(unsigned short));

    // 1) LayerNorm (NCHW f32 -> NHWC bf16)
    ln_kernel<<<(BN * HWP + 255) / 256, 256, 0, stream>>>(x, ln_w, ln_b, xn16);
    // 2) pack weights
    pack_om_kernel<<<(18 * 32 * 32 + 255) / 256, 256, 0, stream>>>(off_w, mask_w, off_b, mask_b, bp_om, bias_om);
    pack64_kernel<<<(18 * 64 * 32 + 255) / 256, 256, 0, stream>>>(def_w, bp_def);
    pack64_kernel<<<(18 * 64 * 32 + 255) / 256, 256, 0, stream>>>(fus_w, bp_fus);
    // 3) offset+mask conv (27 -> padded 32 outputs, f32 NCHW out)
    conv3x3_wmma<2, false><<<NBLK, 32, 0, stream>>>(xn16, bp_om, bias_om, om, pre_mask, x);
    // 4) modulated deformable conv (-> NHWC bf16)
    deform_wmma_kernel<<<NBLK, 32, 0, stream>>>(xn16, om, bp_def, def_b, xdef16);
    // 5) fusion conv + pre_mask blend (-> f32 NCHW d_out)
    conv3x3_wmma<4, true><<<NBLK, 32, 0, stream>>>(xdef16, bp_fus, fus_b, out, pre_mask, x);
}
